// GNN_topk_encode_80212809220208
// MI455X (gfx1250) — matrix operand order repaired and checked by reading, not compiled, not run on hardware
//
#include <hip/hip_runtime.h>

// ---------------------------------------------------------------------------
// MeshGraphNet-style GNN forward for MI455X (gfx1250), bf16 WMMA + f32 LN.
//
// WMMA operand layout used by this code (wave32; same contiguous k order for A and B):
//  A (16x32 bf16): lane<16 -> row M=lane, K = 0..15 ascending;
//                  lane>=16 -> row M=lane-16, K = 16..31 ascending
//  B (32x16 bf16): lane<16 -> col N=lane,    K = 0..15 ascending;
//                  lane>=16 -> col N=lane-16, K = 16..31 ascending
//  C/D (16x16 f32): element v of v8f = row (v + 8*(lane>=16)), col lane&15
// Weights are pre-packed into per-lane B streams so each operand load is one
// contiguous 32B read (L2-resident).
// ---------------------------------------------------------------------------

typedef __attribute__((ext_vector_type(16))) __bf16 v16bf;
typedef __attribute__((ext_vector_type(8)))  float  v8f;

constexpr int Hd  = 128;     // hidden
constexpr int FNd = 6;       // node feats
constexpr int FEd = 3;       // edge feats
constexpr int Nn  = 40000;   // nodes per chunk
constexpr int Ne  = 240000;  // edges
constexpr int Cc  = 2;       // chunks
constexpr int Ss  = 4;       // MP steps

__device__ __forceinline__ unsigned short f2bf(float f) {
  unsigned int u = __float_as_uint(f);
  unsigned int r = u + 0x7FFFu + ((u >> 16) & 1u);   // round-to-nearest-even
  return (unsigned short)(r >> 16);
}

__device__ __forceinline__ float eluf(float x) {
  return x > 0.f ? x : (__expf(x) - 1.f);
}

// ---------------------------------------------------------------------------
// Weight packing: f32 [K][128] row-major -> bf16 per-lane B streams.
// packed[(kt*8+nt)*512 + lane*16 + j] = W[kt*32 + (lane>=16?16:0) + j][nt*16 + (lane&15)]
// ---------------------------------------------------------------------------
__global__ void pack_w_kernel(const float* __restrict__ w,
                              unsigned short* __restrict__ wp, int K) {
  int idx = blockIdx.x * blockDim.x + threadIdx.x;
  int total = K * Hd;
  if (idx >= total) return;
  int tile = idx >> 9;
  int t    = idx & 511;
  int lane = t >> 4;
  int j    = t & 15;
  int kt = tile >> 3;
  int nt = tile & 7;
  int n = nt * 16 + (lane & 15);
  int k = kt * 32 + ((lane >> 4) ? 16 : 0) + j;
  wp[idx] = f2bf(w[k * Hd + n]);
}

__global__ void zero_f32_kernel(float* __restrict__ p, int n) {
  int i = blockIdx.x * blockDim.x + threadIdx.x;
  if (i < n) p[i] = 0.f;
}

__global__ void count_edges_kernel(const int* __restrict__ ei1,
                                   float* __restrict__ cnt) {
  int i = blockIdx.x * blockDim.x + threadIdx.x;
  if (i < Ne) atomicAdd(&cnt[ei1[i]], 1.0f);
}

// ---------------------------------------------------------------------------
// Per-wave GEMM helpers
// ---------------------------------------------------------------------------
__device__ __forceinline__ v16bf load_a_lds(const unsigned short* __restrict__ row,
                                            int kbase, int laneHi) {
  union { v16bf v; unsigned short s[16]; } u;
  int base = kbase + (laneHi ? 16 : 0);
#pragma unroll
  for (int j = 0; j < 16; ++j) u.s[j] = row[base + j];
  return u.v;
}

__device__ __forceinline__ v16bf load_b_glb(const unsigned short* __restrict__ p) {
  union { v16bf v; unsigned short s[16]; } u;
#pragma unroll
  for (int j = 0; j < 16; ++j) u.s[j] = p[j];
  return u.v;
}

// A in LDS: [16][LS] bf16 (row-major, padded stride). W packed B stream.
// acc[nt] covers output cols nt*16..nt*16+15.
template <int KT, int LS>
__device__ __forceinline__ void wave_gemm(const unsigned short* __restrict__ aLds,
                                          const unsigned short* __restrict__ wp,
                                          int lane, v8f acc[8]) {
  int laneHi = lane >> 4;
  const unsigned short* row = aLds + (lane & 15) * LS;
#pragma unroll
  for (int kt = 0; kt < KT; ++kt) {
    v16bf a = load_a_lds(row, kt * 32, laneHi);
#pragma unroll
    for (int nt = 0; nt < 8; ++nt) {
      v16bf b = load_b_glb(wp + ((kt * 8 + nt) * 512) + lane * 16);
      acc[nt] = __builtin_amdgcn_wmma_f32_16x16x32_bf16(
          false, a, false, b, (short)0, acc[nt], false, false);
    }
  }
}

// bias + ELU on D tile, store bf16 to LDS [16][136] (A operand for next GEMM)
__device__ __forceinline__ void elu_store_a2(const v8f acc[8],
                                             const float* __restrict__ b,
                                             unsigned short* __restrict__ a2,
                                             int lane) {
  int laneHi = lane >> 4, ln = lane & 15;
#pragma unroll
  for (int nt = 0; nt < 8; ++nt) {
    int col = nt * 16 + ln;
    float bb = b[col];
#pragma unroll
    for (int v = 0; v < 8; ++v) {
      float x = acc[nt][v] + bb;
      a2[(v + 8 * laneHi) * 136 + col] = f2bf(eluf(x));
    }
  }
}

// bias, store D tile f32 to LDS [16][132]
__device__ __forceinline__ void store_d(const v8f acc[8],
                                        const float* __restrict__ b,
                                        float* __restrict__ dbuf, int lane) {
  int laneHi = lane >> 4, ln = lane & 15;
#pragma unroll
  for (int nt = 0; nt < 8; ++nt) {
    float bb = b[nt * 16 + ln];
#pragma unroll
    for (int v = 0; v < 8; ++v)
      dbuf[(v + 8 * laneHi) * 132 + nt * 16 + ln] = acc[nt][v] + bb;
  }
}

// ---------------------------------------------------------------------------
// Encoder: norm -> L0(scalar, tiny K)+ELU -> L1(WMMA)+ELU -> L2(WMMA) -> LN
// One 16-row tile per wave, 4 waves/block.
// ---------------------------------------------------------------------------
constexpr int ENCREG = 8448;
__global__ __launch_bounds__(128) void encoder_kernel(
    const float* __restrict__ x, int F,
    const float* __restrict__ meanv, const float* __restrict__ stdv,
    const float* __restrict__ w0, const float* __restrict__ b0,
    const unsigned short* __restrict__ w1p, const float* __restrict__ b1,
    const unsigned short* __restrict__ w2p, const float* __restrict__ b2,
    const float* __restrict__ g, const float* __restrict__ be,
    float* __restrict__ out) {
  __shared__ alignas(16) char smem[4 * ENCREG];
  int wave = threadIdx.x >> 5, lane = threadIdx.x & 31;
  char* my = smem + wave * ENCREG;
  unsigned short* a1 = (unsigned short*)my;  // [16][136] bf16
  float* dbuf = (float*)my;                  // [16][132] f32
  int r0 = (blockIdx.x * 4 + wave) * 16;

  // layer 0 (K = 3 or 6): scalar VALU, cooperative over lanes
  for (int r = 0; r < 16; ++r) {
    const float* xr = x + (size_t)(r0 + r) * F;
    float xn[FNd];
    for (int k = 0; k < F; ++k) xn[k] = (xr[k] - meanv[k]) / stdv[k];
    unsigned short* row = a1 + r * 136;
    for (int c = lane; c < Hd; c += 32) {
      float s = b0[c];
      for (int k = 0; k < F; ++k) s += xn[k] * w0[k * Hd + c];
      row[c] = f2bf(eluf(s));
    }
  }
  __syncthreads();
  v8f acc[8] = {};
  wave_gemm<4, 136>(a1, w1p, lane, acc);
  __syncthreads();
  elu_store_a2(acc, b1, a1, lane);   // reuse region as A2
  __syncthreads();
  v8f acc2[8] = {};
  wave_gemm<4, 136>(a1, w2p, lane, acc2);
  __syncthreads();
  store_d(acc2, b2, dbuf, lane);
  __syncthreads();
  // LayerNorm: 2 lanes per row
  int r = lane & 15, half = lane >> 4;
  float* drow = dbuf + r * 132;
  float s1 = 0.f, s2 = 0.f;
  for (int c = half * 64; c < half * 64 + 64; ++c) {
    float v = drow[c]; s1 += v; s2 += v * v;
  }
  drow[128 + half] = s1;
  drow[130 + half] = s2;
  __syncthreads();
  float mean = (drow[128] + drow[129]) * (1.f / Hd);
  float var  = (drow[130] + drow[131]) * (1.f / Hd) - mean * mean;
  float rstd = rsqrtf(var + 1e-5f);
  float* orow = out + (size_t)(r0 + r) * Hd;
  for (int c = half * 64; c < half * 64 + 64; ++c)
    orow[c] = (drow[c] - mean) * rstd * g[c] + be[c];
}

// ---------------------------------------------------------------------------
// Edge MP step: gather [h[src]|h[dst]|e] -> MLP(384->128->128) -> +e -> LN
//               -> write e, atomic scatter-add to agg[target]
// ---------------------------------------------------------------------------
constexpr int EREG = 13312;  // >= max(16*408*2, 16*132*4)
__global__ __launch_bounds__(128) void edge_mp_kernel(
    const float* __restrict__ h, const float* e_in, float* e_out,
    float* __restrict__ agg,
    const int* __restrict__ ei0, const int* __restrict__ ei1,
    const unsigned short* __restrict__ w0p, const float* __restrict__ b0,
    const unsigned short* __restrict__ w1p, const float* __restrict__ b1,
    const float* __restrict__ g, const float* __restrict__ be) {
  __shared__ alignas(16) char smem[4 * EREG];
  int wave = threadIdx.x >> 5, lane = threadIdx.x & 31;
  char* my = smem + wave * EREG;
  unsigned short* a1 = (unsigned short*)my;  // [16][408] bf16 (concat K=384)
  float* dbuf = (float*)my;                  // [16][132] f32
  int e0i = (blockIdx.x * 4 + wave) * 16;

  // stage concat tile (coalesced 32-lane reads per row)
  for (int r = 0; r < 16; ++r) {
    int eIdx = e0i + r;
    const float* ph0 = h + (size_t)ei0[eIdx] * Hd;
    const float* ph1 = h + (size_t)ei1[eIdx] * Hd;
    const float* pe  = e_in + (size_t)eIdx * Hd;
    unsigned short* row = a1 + r * 408;
    for (int c = lane; c < Hd; c += 32) {
      row[c]          = f2bf(ph0[c]);
      row[Hd + c]     = f2bf(ph1[c]);
      row[2 * Hd + c] = f2bf(pe[c]);
    }
  }
  __syncthreads();
  v8f acc[8] = {};
  wave_gemm<12, 408>(a1, w0p, lane, acc);   // 96 wmma
  __syncthreads();
  unsigned short* a2 = (unsigned short*)my;  // [16][136]
  elu_store_a2(acc, b0, a2, lane);
  __syncthreads();
  v8f acc2[8] = {};
  wave_gemm<4, 136>(a2, w1p, lane, acc2);   // 32 wmma
  __syncthreads();
  store_d(acc2, b1, dbuf, lane);
  __syncthreads();
  // residual + LN + scatter (2 lanes per row)
  int r = lane & 15, half = lane >> 4;
  int eIdx = e0i + r;
  const float* ein_row = e_in + (size_t)eIdx * Hd;
  float* drow = dbuf + r * 132;
  float s1 = 0.f, s2 = 0.f;
  for (int c = half * 64; c < half * 64 + 64; ++c) {
    float v = drow[c] + ein_row[c];
    drow[c] = v; s1 += v; s2 += v * v;
  }
  drow[128 + half] = s1;
  drow[130 + half] = s2;
  __syncthreads();
  float mean = (drow[128] + drow[129]) * (1.f / Hd);
  float var  = (drow[130] + drow[131]) * (1.f / Hd) - mean * mean;
  float rstd = rsqrtf(var + 1e-5f);
  float* aggrow  = agg + (size_t)ei1[eIdx] * Hd;
  float* eout_row = e_out + (size_t)eIdx * Hd;
  for (int c = half * 64; c < half * 64 + 64; ++c) {
    float v = (drow[c] - mean) * rstd * g[c] + be[c];
    eout_row[c] = v;
    atomicAdd(&aggrow[c], v);
  }
}

// ---------------------------------------------------------------------------
// Node MP step: [h | agg/cnt] -> MLP(256->128->128) -> +h -> LN -> h
// ---------------------------------------------------------------------------
constexpr int NREG = 8448;  // >= max(16*264*2, 16*132*4)
__global__ __launch_bounds__(128) void node_mp_kernel(
    float* __restrict__ h, const float* __restrict__ agg,
    const float* __restrict__ cnt,
    const unsigned short* __restrict__ w0p, const float* __restrict__ b0,
    const unsigned short* __restrict__ w1p, const float* __restrict__ b1,
    const float* __restrict__ g, const float* __restrict__ be) {
  __shared__ alignas(16) char smem[4 * NREG];
  int wave = threadIdx.x >> 5, lane = threadIdx.x & 31;
  char* my = smem + wave * NREG;
  unsigned short* a1 = (unsigned short*)my;  // [16][264] bf16 (concat K=256)
  float* dbuf = (float*)my;                  // [16][132] f32
  int n0 = (blockIdx.x * 4 + wave) * 16;

  for (int r = 0; r < 16; ++r) {
    int node = n0 + r;
    const float* ph = h + (size_t)node * Hd;
    const float* pa = agg + (size_t)node * Hd;
    float rc = 1.f / fmaxf(cnt[node], 1.f);
    unsigned short* row = a1 + r * 264;
    for (int c = lane; c < Hd; c += 32) {
      row[c]      = f2bf(ph[c]);
      row[Hd + c] = f2bf(pa[c] * rc);
    }
  }
  __syncthreads();
  v8f acc[8] = {};
  wave_gemm<8, 264>(a1, w0p, lane, acc);
  __syncthreads();
  unsigned short* a2 = (unsigned short*)my;
  elu_store_a2(acc, b0, a2, lane);
  __syncthreads();
  v8f acc2[8] = {};
  wave_gemm<4, 136>(a2, w1p, lane, acc2);
  __syncthreads();
  store_d(acc2, b1, dbuf, lane);
  __syncthreads();
  int r = lane & 15, half = lane >> 4;
  int node = n0 + r;
  float* hrow = h + (size_t)node * Hd;
  float* drow = dbuf + r * 132;
  float s1 = 0.f, s2 = 0.f;
  for (int c = half * 64; c < half * 64 + 64; ++c) {
    float v = drow[c] + hrow[c];
    drow[c] = v; s1 += v; s2 += v * v;
  }
  drow[128 + half] = s1;
  drow[130 + half] = s2;
  __syncthreads();
  float mean = (drow[128] + drow[129]) * (1.f / Hd);
  float var  = (drow[130] + drow[131]) * (1.f / Hd) - mean * mean;
  float rstd = rsqrtf(var + 1e-5f);
  for (int c = half * 64; c < half * 64 + 64; ++c)
    hrow[c] = (drow[c] - mean) * rstd * g[c] + be[c];
}

// ---------------------------------------------------------------------------
extern "C" void kernel_launch(void* const* d_in, const int* in_sizes, int n_in,
                              void* d_out, int out_size, void* d_ws, size_t ws_size,
                              hipStream_t stream) {
  (void)in_sizes; (void)n_in; (void)out_size; (void)ws_size;
  const float* x         = (const float*)d_in[0];
  const float* edge_attr = (const float*)d_in[1];
  const float* mean_x = (const float*)d_in[2];
  const float* std_x  = (const float*)d_in[3];
  const float* mean_e = (const float*)d_in[4];
  const float* std_e  = (const float*)d_in[5];
  const float* nenc_w0 = (const float*)d_in[6];  const float* nenc_b0 = (const float*)d_in[7];
  const float* nenc_w1 = (const float*)d_in[8];  const float* nenc_b1 = (const float*)d_in[9];
  const float* nenc_w2 = (const float*)d_in[10]; const float* nenc_b2 = (const float*)d_in[11];
  const float* nenc_g  = (const float*)d_in[12]; const float* nenc_be = (const float*)d_in[13];
  const float* eenc_w0 = (const float*)d_in[14]; const float* eenc_b0 = (const float*)d_in[15];
  const float* eenc_w1 = (const float*)d_in[16]; const float* eenc_b1 = (const float*)d_in[17];
  const float* eenc_w2 = (const float*)d_in[18]; const float* eenc_b2 = (const float*)d_in[19];
  const float* eenc_g  = (const float*)d_in[20]; const float* eenc_be = (const float*)d_in[21];
  const float* emp_w0 = (const float*)d_in[22];  const float* emp_b0 = (const float*)d_in[23];
  const float* emp_w1 = (const float*)d_in[24];  const float* emp_b1 = (const float*)d_in[25];
  const float* emp_g  = (const float*)d_in[26];  const float* emp_be = (const float*)d_in[27];
  const float* nmp_w0 = (const float*)d_in[28];  const float* nmp_b0 = (const float*)d_in[29];
  const float* nmp_w1 = (const float*)d_in[30];  const float* nmp_b1 = (const float*)d_in[31];
  const float* nmp_g  = (const float*)d_in[32];  const float* nmp_be = (const float*)d_in[33];
  const int* ei  = (const int*)d_in[34];
  const int* ei0 = ei;
  const int* ei1 = ei + Ne;

  // workspace carve-up
  char* ws = (char*)d_ws;
  float* e   = (float*)ws; ws += (size_t)Ne * Hd * sizeof(float);
  float* agg = (float*)ws; ws += (size_t)Nn * Hd * sizeof(float);
  float* cnt = (float*)ws; ws += (size_t)Nn * sizeof(float);
  unsigned short* pw = (unsigned short*)ws;
  unsigned short* p_ew1 = pw;
  unsigned short* p_ew2 = p_ew1 + 16384;
  unsigned short* p_nw1 = p_ew2 + 16384;
  unsigned short* p_nw2 = p_nw1 + 16384;
  unsigned short* p_empw0 = p_nw2 + 16384;                       // S*384*128
  unsigned short* p_empw1 = p_empw0 + (size_t)Ss * 3 * Hd * Hd;  // S*128*128
  unsigned short* p_nmpw0 = p_empw1 + (size_t)Ss * Hd * Hd;      // S*256*128
  unsigned short* p_nmpw1 = p_nmpw0 + (size_t)Ss * 2 * Hd * Hd;  // S*128*128

  auto packB = [&](const float* w, unsigned short* dst, int K) {
    int tot = K * Hd;
    pack_w_kernel<<<(tot + 255) / 256, 256, 0, stream>>>(w, dst, K);
  };
  packB(eenc_w1, p_ew1, Hd);
  packB(eenc_w2, p_ew2, Hd);
  packB(nenc_w1, p_nw1, Hd);
  packB(nenc_w2, p_nw2, Hd);
  for (int s = 0; s < Ss; ++s) {
    packB(emp_w0 + (size_t)s * 3 * Hd * Hd, p_empw0 + (size_t)s * 3 * Hd * Hd, 3 * Hd);
    packB(emp_w1 + (size_t)s * Hd * Hd,     p_empw1 + (size_t)s * Hd * Hd,     Hd);
    packB(nmp_w0 + (size_t)s * 2 * Hd * Hd, p_nmpw0 + (size_t)s * 2 * Hd * Hd, 2 * Hd);
    packB(nmp_w1 + (size_t)s * Hd * Hd,     p_nmpw1 + (size_t)s * Hd * Hd,     Hd);
  }

  // scatter-mean denominator
  zero_f32_kernel<<<(Nn + 255) / 256, 256, 0, stream>>>(cnt, Nn);
  count_edges_kernel<<<(Ne + 255) / 256, 256, 0, stream>>>(ei1, cnt);

  // node encoder over all C*N rows, directly into d_out (== h)
  float* hall = (float*)d_out;
  encoder_kernel<<<(Cc * Nn / 16) / 4, 128, 0, stream>>>(
      x, FNd, mean_x, std_x, nenc_w0, nenc_b0, p_nw1, nenc_b1, p_nw2, nenc_b2,
      nenc_g, nenc_be, hall);

  for (int c = 0; c < Cc; ++c) {
    // fresh edge encoding per chunk (cheap, L2-bound; avoids buffering e0)
    encoder_kernel<<<(Ne / 16) / 4, 128, 0, stream>>>(
        edge_attr, FEd, mean_e, std_e, eenc_w0, eenc_b0, p_ew1, eenc_b1,
        p_ew2, eenc_b2, eenc_g, eenc_be, e);
    float* hc = hall + (size_t)c * Nn * Hd;
    for (int s = 0; s < Ss; ++s) {
      zero_f32_kernel<<<(Nn * Hd + 255) / 256, 256, 0, stream>>>(agg, Nn * Hd);
      edge_mp_kernel<<<(Ne / 16) / 4, 128, 0, stream>>>(
          hc, e, e, agg, ei0, ei1,
          p_empw0 + (size_t)s * 3 * Hd * Hd, emp_b0 + s * Hd,
          p_empw1 + (size_t)s * Hd * Hd,     emp_b1 + s * Hd,
          emp_g + s * Hd, emp_be + s * Hd);
      node_mp_kernel<<<(Nn / 16) / 4, 128, 0, stream>>>(
          hc, agg, cnt,
          p_nmpw0 + (size_t)s * 2 * Hd * Hd, nmp_b0 + s * Hd,
          p_nmpw1 + (size_t)s * Hd * Hd,     nmp_b1 + s * Hd,
          nmp_g + s * Hd, nmp_be + s * Hd);
    }
  }
}
